// DAM2_68934225101109
// MI455X (gfx1250) — compile-verified, matrix operand order checked by reading, not run on hardware
//
#include <hip/hip_runtime.h>
#include <hip/hip_bf16.h>

// ---------- problem constants ----------
#define BB   8
#define CC   256
#define HH   128
#define WW   128
#define HWD  (HH * WW)          // 16384
#define PTOT (BB * HWD)         // 131072 pixels
#define PLANE ((size_t)BB * CC * HWD)  // one [B,C,H,W] plane in floats
#define PADW 134
#define PADSZ (BB * PADW * PADW)

// ---------- vector types ----------
typedef __attribute__((ext_vector_type(16))) __bf16         v16bf;
typedef __attribute__((ext_vector_type(8)))  float          v8f;
typedef __attribute__((ext_vector_type(16))) unsigned short v16u;
typedef __attribute__((ext_vector_type(8)))  unsigned short v8u;
typedef __attribute__((ext_vector_type(4)))  unsigned int   u32x4;
typedef __attribute__((ext_vector_type(8)))  unsigned int   u32x8;

union BFrag {            // one 16x32 (or 32x16) bf16 WMMA operand
    v16bf b;
    v16u  u;
    v8u   h[2];
};

__device__ __forceinline__ unsigned short f2bf(float f) {
    unsigned int u = __float_as_uint(f);
    u += 0x7FFFu + ((u >> 16) & 1u);     // round-to-nearest-even
    return (unsigned short)(u >> 16);
}

__device__ __forceinline__ unsigned int pack2(float lo, float hi) {
    return (unsigned int)f2bf(lo) | ((unsigned int)f2bf(hi) << 16);
}

__device__ __forceinline__ v8f wmma_bf16(const BFrag& a, const BFrag& b, v8f c) {
    return __builtin_amdgcn_wmma_f32_16x16x32_bf16(false, a.b, false, b.b,
                                                   (short)0, c, false, false);
}

// ---------------------------------------------------------------------
// Tensor Data Mover: async 2D tile load (global -> LDS), 2-group D#.
// g0 = {count=1 | lds_addr | global_addr(57b) | type=2}; g1 carries
// data_size, optional LDS padding, tensor dims (huge -> no OOB),
// tile dims, dim0 stride. Completion tracked with TENSORcnt.
// ---------------------------------------------------------------------
__device__ __forceinline__ u32x4 tdm_g0(unsigned lds_off, const void* gptr) {
    unsigned long long ga = (unsigned long long)(size_t)gptr;
    u32x4 g0;
    g0[0] = 1u;                                       // count=1, user descriptor
    g0[1] = lds_off;                                  // lds_addr (bytes)
    g0[2] = (unsigned)(ga & 0xFFFFFFFFu);             // global_addr[31:0]
    g0[3] = (unsigned)((ga >> 32) & 0x01FFFFFFu)      // global_addr[56:32]
          | (2u << 30);                               // type=2 ("image")
    return g0;
}

// data_size: 0=1B 1=2B 2=4B; pad_interval: 2^(pi+1) DW; pad_amount: pa+1 DW
__device__ __forceinline__ u32x8 tdm_g1x(unsigned tile0, unsigned tile1,
                                         unsigned long long stride_elems,
                                         unsigned data_size, unsigned pad_en,
                                         unsigned pad_interval, unsigned pad_amount) {
    const unsigned td0 = 0x01000000u;                 // tensor_dim0 (huge)
    const unsigned td1 = 0x00100000u;                 // tensor_dim1 (huge)
    u32x8 g1;
    g1[0] = (data_size << 16) | (pad_en << 20)
          | (pad_interval << 22) | (pad_amount << 25);
    g1[1] = (td0 & 0xFFFFu) << 16;                    // atomic_addr=0 | dim0 lo16
    g1[2] = ((td0 >> 16) & 0xFFFFu) | ((td1 & 0xFFFFu) << 16);
    g1[3] = ((td1 >> 16) & 0xFFFFu) | (tile0 << 16);  // dim1 hi16 | tile_dim0
    g1[4] = (tile1 & 0xFFFFu);                        // tile_dim1 | tile_dim2=0
    g1[5] = (unsigned)(stride_elems & 0xFFFFFFFFu);   // dim0_stride[31:0]
    g1[6] = (unsigned)((stride_elems >> 32) & 0xFFFFu);
    g1[7] = 0u;
    return g1;
}

__device__ __forceinline__ void tdm_issue(u32x4 g0, u32x8 g1) {
    asm volatile("tensor_load_to_lds %0, %1" :: "s"(g0), "s"(g1) : "memory");
}

// =====================================================================
// 3x3 avg (count_include_pad -> /9) + 3x3 max (pad = -inf) in one pass
// =====================================================================
__global__ __launch_bounds__(256)
void pool3_kernel(const float* __restrict__ x, float* __restrict__ yavg,
                  float* __restrict__ ymax) {
    size_t idx = (size_t)blockIdx.x * 256 + threadIdx.x;
    if (idx >= PLANE) return;
    int w = (int)(idx % WW);
    int h = (int)((idx / WW) % HH);
    size_t bc = idx / HWD;
    const float* img = x + bc * HWD;
    float s = 0.f, m = -INFINITY;
    #pragma unroll
    for (int dy = -1; dy <= 1; ++dy)
        #pragma unroll
        for (int dx = -1; dx <= 1; ++dx) {
            int yy = h + dy, xx = w + dx;
            if (yy >= 0 && yy < HH && xx >= 0 && xx < WW) {
                float v = img[yy * WW + xx];
                s += v;
                m = fmaxf(m, v);
            }
        }
    yavg[idx] = s * (1.f / 9.f);
    ymax[idx] = m;
}

// =====================================================================
// bf16-WMMA GEMM for 1x1 convs, TDM-staged, double-buffered, bf16 weights:
//   Y[b,m,p] = act( sum_k W[m,k]*X[b,k,p] + bias[m] (+addsrc) (+wv[m]*vvec) )
// One workgroup owns ALL M=256 output channels for 64 pixels; each X tile
// is DMA'd + converted exactly once. 8 waves x (32m x 64n) block.
// Weights are pre-converted bf16 row-major: A frag = two global_load_b128.
// act (omode 0): 0 = none, 1 = relu, 2 = sigmoid * gate[b,m,p]
// omode: 0 = f32 [b][m][p]; 1 = bf16 transposed [b][p][m] (one b128 store
//        per tile-half); 2 = bf16 normal [b][m][p]
// =====================================================================
#define XS_STRIDE 40    // ushorts per LDS n-row (16B-aligned 8-K runs)

__global__ __launch_bounds__(256)
void gemm1x1_kernel(const unsigned short* __restrict__ Wb, int ldw,
                    const float* __restrict__ bias,
                    const float* __restrict__ Xa,
                    const float* __restrict__ Xb,
                    int Ka, int Ktot,
                    void* __restrict__ Yv,
                    const float* __restrict__ addsrc,
                    const float* __restrict__ gate,
                    const float* __restrict__ vvec,
                    const float* __restrict__ wv,
                    int M, int act, int omode) {
    __shared__ float raws[2][32 * 64];            // TDM double buffer (raw f32)
    __shared__ unsigned short xs[64 * XS_STRIDE]; // bf16, transposed [n][k]

    const int tid  = threadIdx.x;
    const int lane = tid & 31;
    const int wave = tid >> 5;
    const int l15  = lane & 15;
    const int ko   = (lane < 16) ? 0 : 8;

    const int nblk = blockIdx.x * 64;
    const int b    = nblk / HWD;        // 64-col tile never crosses a batch
    const int p0   = nblk % HWD;

    const int nk = Ktot >> 5;           // K-steps of 32

    // conversion mapping: thread -> (k pair, 4 n's)
    const int kp = (tid & 15) * 2;
    const int nb = (tid >> 4) * 4;

    v8f acc[2][4] = {};

    const u32x8 g1t = tdm_g1x(64u, 32u, (unsigned long long)HWD, 2u, 0u, 0u, 0u);

    // prologue: kick off first DMA
    if (wave == 0) {
        tdm_issue(tdm_g0((unsigned)(size_t)&raws[0][0],
                         Xa + (size_t)b * Ka * HWD + p0), g1t);
    }

    for (int i = 0; i < nk; ++i) {
        const int cur = i & 1;
        if (wave == 0) __builtin_amdgcn_s_wait_tensorcnt(0);
        __syncthreads();                          // raws[cur] ready for all waves
        if (wave == 0 && i + 1 < nk) {            // overlap next DMA with compute
            const int k1 = (i + 1) * 32;
            const float* src = (k1 < Ka)
                ? Xa + (size_t)b * Ka * HWD + (size_t)k1 * HWD + p0
                : Xb + (size_t)b * (Ktot - Ka) * HWD + (size_t)(k1 - Ka) * HWD + p0;
            tdm_issue(tdm_g0((unsigned)(size_t)&raws[cur ^ 1][0], src), g1t);
        }

        // ---- convert raws[cur] (f32 [k][n]) -> xs (bf16 [n][k]) ----
        {
            const float* r0 = &raws[cur][kp * 64 + nb];
            const float* r1 = &raws[cur][(kp + 1) * 64 + nb];
            float4 a = *(const float4*)r0;
            float4 c = *(const float4*)r1;
            unsigned int* xs32 = (unsigned int*)xs;
            const int kw = kp >> 1;
            xs32[(nb + 0) * 20 + kw] = pack2(a.x, c.x);
            xs32[(nb + 1) * 20 + kw] = pack2(a.y, c.y);
            xs32[(nb + 2) * 20 + kw] = pack2(a.z, c.z);
            xs32[(nb + 3) * 20 + kw] = pack2(a.w, c.w);
        }
        __syncthreads();                          // xs ready

        // ---- A fragments: bf16 weights, two global_load_b128 each ----
        const int k0 = i * 32;
        BFrag af[2];
        #pragma unroll
        for (int mt = 0; mt < 2; ++mt) {
            const int m = wave * 32 + mt * 16 + l15;
            const unsigned short* wr = Wb + (size_t)m * ldw + k0 + ko;
            af[mt].h[0] = *(const v8u*)(wr);
            af[mt].h[1] = *(const v8u*)(wr + 16);
        }

        // ---- B fragments from LDS (two ds_load_b128 each), 8 WMMAs ----
        #pragma unroll
        for (int jt = 0; jt < 4; ++jt) {
            BFrag bf;
            const unsigned short* bp = &xs[(jt * 16 + l15) * XS_STRIDE + ko];
            bf.h[0] = *(const v8u*)(bp);
            bf.h[1] = *(const v8u*)(bp + 16);
            acc[0][jt] = wmma_bf16(af[0], bf, acc[0][jt]);
            acc[1][jt] = wmma_bf16(af[1], bf, acc[1][jt]);
        }
        __syncthreads();                          // xs consumed
    }

    // ---- epilogue ----
    if (omode == 1) {
        // bf16 transposed [b][p][m]: 8 consecutive channels per lane -> b128
        unsigned short* Yt = (unsigned short*)Yv;
        #pragma unroll
        for (int mt = 0; mt < 2; ++mt)
            #pragma unroll
            for (int jt = 0; jt < 4; ++jt) {
                const int p = p0 + jt * 16 + l15;
                const int mbase = wave * 32 + mt * 16 + ((lane >= 16) ? 8 : 0);
                float v0 = acc[mt][jt][0] + bias[mbase + 0];
                float v1 = acc[mt][jt][1] + bias[mbase + 1];
                float v2 = acc[mt][jt][2] + bias[mbase + 2];
                float v3 = acc[mt][jt][3] + bias[mbase + 3];
                float v4 = acc[mt][jt][4] + bias[mbase + 4];
                float v5 = acc[mt][jt][5] + bias[mbase + 5];
                float v6 = acc[mt][jt][6] + bias[mbase + 6];
                float v7 = acc[mt][jt][7] + bias[mbase + 7];
                u32x4 pk;
                pk[0] = pack2(v0, v1);
                pk[1] = pack2(v2, v3);
                pk[2] = pack2(v4, v5);
                pk[3] = pack2(v6, v7);
                *(u32x4*)(Yt + ((size_t)b * HWD + p) * M + mbase) = pk;
            }
    } else if (omode == 2) {
        // bf16 normal [b][m][p]
        unsigned short* Yb = (unsigned short*)Yv;
        #pragma unroll
        for (int mt = 0; mt < 2; ++mt)
            #pragma unroll
            for (int jt = 0; jt < 4; ++jt) {
                const int p = p0 + jt * 16 + l15;
                #pragma unroll
                for (int v = 0; v < 8; ++v) {
                    const int m = wave * 32 + mt * 16 + v + ((lane >= 16) ? 8 : 0);
                    Yb[((size_t)b * M + m) * HWD + p] = f2bf(acc[mt][jt][v] + bias[m]);
                }
            }
    } else {
        float* Y = (float*)Yv;
        #pragma unroll
        for (int mt = 0; mt < 2; ++mt)
            #pragma unroll
            for (int jt = 0; jt < 4; ++jt) {
                const int p = p0 + jt * 16 + l15;
                #pragma unroll
                for (int v = 0; v < 8; ++v) {
                    const int m = wave * 32 + mt * 16 + v + ((lane >= 16) ? 8 : 0);
                    const size_t oidx = ((size_t)b * M + m) * HWD + p;
                    float val = acc[mt][jt][v];
                    if (bias)   val += bias[m];
                    if (addsrc) val += addsrc[oidx];
                    if (wv)     val += wv[m] * vvec[(size_t)b * HWD + p];
                    if (act == 1) {
                        val = fmaxf(val, 0.f);
                    } else if (act == 2) {
                        val = gate[oidx] / (1.f + __expf(-val));
                    }
                    Y[oidx] = val;
                }
            }
    }
}

// =====================================================================
// CA layer: HW mean+max reduction, tiny MLP, per-channel gating
// =====================================================================
__global__ __launch_bounds__(256)
void ca_reduce_kernel(const float* __restrict__ xc, float* __restrict__ mean,
                      float* __restrict__ mx) {
    const int bc = blockIdx.x;
    const float* p = xc + (size_t)bc * HWD;
    float s = 0.f, m = -INFINITY;
    for (int i = threadIdx.x; i < HWD; i += 256) {
        float v = p[i];
        s += v;
        m = fmaxf(m, v);
    }
    __shared__ float ss[256];
    __shared__ float sm[256];
    ss[threadIdx.x] = s; sm[threadIdx.x] = m;
    __syncthreads();
    for (int off = 128; off > 0; off >>= 1) {
        if (threadIdx.x < off) {
            ss[threadIdx.x] += ss[threadIdx.x + off];
            sm[threadIdx.x]  = fmaxf(sm[threadIdx.x], sm[threadIdx.x + off]);
        }
        __syncthreads();
    }
    if (threadIdx.x == 0) {
        mean[bc] = ss[0] / (float)HWD;
        mx[bc]   = sm[0];
    }
}

__global__ __launch_bounds__(256)
void ca_mlp_kernel(const float* __restrict__ mean, const float* __restrict__ mx,
                   const float* aw1, const float* ab1, const float* aw2, const float* ab2,
                   const float* mw1, const float* mb1, const float* mw2, const float* mb2,
                   float* __restrict__ scale) {
    const int b = blockIdx.x, tid = threadIdx.x;
    __shared__ float ha[16], hm[16];
    if (tid < 16) {
        float sa = ab1[tid], sm2 = mb1[tid];
        for (int c = 0; c < CC; ++c) {
            sa  += aw1[tid * CC + c] * mean[b * CC + c];
            sm2 += mw1[tid * CC + c] * mx[b * CC + c];
        }
        ha[tid] = fmaxf(sa, 0.f);
        hm[tid] = fmaxf(sm2, 0.f);
    }
    __syncthreads();
    if (tid < CC) {
        float ya = ab2[tid], ym = mb2[tid];
        #pragma unroll
        for (int j = 0; j < 16; ++j) {
            ya += aw2[tid * 16 + j] * ha[j];
            ym += mw2[tid * 16 + j] * hm[j];
        }
        scale[b * CC + tid] = 1.f / (1.f + __expf(-(ya + ym)));
    }
}

__global__ __launch_bounds__(256)
void bufc_kernel(const float* __restrict__ xc, const float* __restrict__ scale,
                 float* __restrict__ out) {
    size_t idx = (size_t)blockIdx.x * 256 + threadIdx.x;
    if (idx >= PLANE) return;
    out[idx] = scale[idx / HWD] * xc[idx];
}

// =====================================================================
// Fused width-attention. One workgroup per (b,h).
// Q/S planes arrive as bf16 TRANSPOSED [b][p][c] (written that way by the
// projection GEMMs), so each (b,h) tile is one contiguous 64KB block.
// One tensor_load_to_lds per operand deposits it at 264-ushort row stride
// using the TDM pad feature (pad_interval = 128 DW = one 256-elem row,
// pad_amount = 4 DW) -- zero conversion/transpose code.
// Phase 1: scores[i,j] = sum_c Q[i,c]*S[j,c] via WMMA (K=256)
// Phase 2: row softmax in LDS (scores f32 overlay the Q-tile region)
// mask kernel : column-sum threshold -> V
// apply kernel: bf16 v3 tile (256x128, normal layout) DMA'd during
//   phase 1+softmax; scores repacked to bf16;
//   Out[c,i] = sum_j v3[c,j]*M[i,j] via WMMA (K=128)
// LDS: qs (67584) | ssm (67584) | vt (65536) = 200704 bytes
// =====================================================================
#define ATT_STRIDE 264
#define SB_STRIDE  136
#define ATT_QS_BYTES (128 * ATT_STRIDE * 2)
#define ATT_LDS_BYTES (2 * ATT_QS_BYTES + 256 * 128 * 2)

__device__ __forceinline__ void attn_scores_softmax(unsigned short* qs,
                                                    unsigned short* ssm,
                                                    float* sc, int tid) {
    const int lane = tid & 31;
    const int wave = tid >> 5;
    const int l15  = lane & 15;
    const int ko   = (lane < 16) ? 0 : 8;
    const int i0   = wave * 16;

    v8f accs[8] = {};
    for (int c0 = 0; c0 < CC; c0 += 32) {
        BFrag af;
        const unsigned short* qp = qs + (i0 + l15) * ATT_STRIDE + c0 + ko;
        af.h[0] = *(const v8u*)(qp);
        af.h[1] = *(const v8u*)(qp + 16);
        #pragma unroll
        for (int jt = 0; jt < 8; ++jt) {
            BFrag bf;
            const unsigned short* sp = ssm + (jt * 16 + l15) * ATT_STRIDE + c0 + ko;
            bf.h[0] = *(const v8u*)(sp);
            bf.h[1] = *(const v8u*)(sp + 16);
            accs[jt] = wmma_bf16(af, bf, accs[jt]);
        }
    }
    __syncthreads();   // everyone done reading qs before scores overlay it
    #pragma unroll
    for (int jt = 0; jt < 8; ++jt)
        #pragma unroll
        for (int v = 0; v < 8; ++v) {
            int i = i0 + v + ((lane >= 16) ? 8 : 0);
            int j = jt * 16 + l15;
            sc[i * WW + j] = accs[jt][v];
        }
    __syncthreads();
    if (tid < WW) {                      // row softmax
        float* row = sc + tid * WW;
        float mxv = -INFINITY;
        for (int j = 0; j < WW; ++j) mxv = fmaxf(mxv, row[j]);
        float sum = 0.f;
        for (int j = 0; j < WW; ++j) { float e = __expf(row[j] - mxv); row[j] = e; sum += e; }
        float inv = 1.f / sum;
        for (int j = 0; j < WW; ++j) row[j] *= inv;
    }
    __syncthreads();
}

// D# for a Q/S tile: 128 rows x 256 bf16, contiguous rows, padded to 264
#define G1_QS tdm_g1x(256u, 128u, 256ull, 1u, 1u, 6u, 3u)

__global__ __launch_bounds__(256)
void attn_mask_kernel(const unsigned short* __restrict__ Q2t,
                      const unsigned short* __restrict__ S2t,
                      float* __restrict__ Vraw) {
    extern __shared__ unsigned char smem[];
    unsigned short* qs  = (unsigned short*)smem;
    unsigned short* ssm = qs + 128 * ATT_STRIDE;
    float* sc = (float*)smem;
    const int b = blockIdx.x / HH, h = blockIdx.x % HH, hoff = h * WW;
    const int tid = threadIdx.x, wave = tid >> 5;

    if (wave == 0) {
        const u32x8 g1 = G1_QS;
        tdm_issue(tdm_g0(0u, Q2t + ((size_t)b * HWD + hoff) * CC), g1);
        tdm_issue(tdm_g0(ATT_QS_BYTES, S2t + ((size_t)b * HWD + hoff) * CC), g1);
        __builtin_amdgcn_s_wait_tensorcnt(0);
    }
    __syncthreads();

    attn_scores_softmax(qs, ssm, sc, tid);

    if (tid < WW) {                      // sum over i (axis=2), threshold
        float s = 0.f;
        for (int i = 0; i < WW; ++i) s += sc[i * WW + tid];
        Vraw[(size_t)b * HWD + hoff + tid] = (s > 0.1f) ? 1.f : 0.f;
    }
}

__global__ __launch_bounds__(256)
void attn_apply_kernel(const unsigned short* __restrict__ Q1t,
                       const unsigned short* __restrict__ S1t,
                       const unsigned short* __restrict__ V3b,
                       float* __restrict__ Out) {
    extern __shared__ unsigned char smem[];
    unsigned short* qs  = (unsigned short*)smem;
    unsigned short* ssm = qs + 128 * ATT_STRIDE;
    unsigned short* vt  = ssm + 128 * ATT_STRIDE;
    float* sc = (float*)smem;
    const int b = blockIdx.x / HH, h = blockIdx.x % HH, hoff = h * WW;
    const int tid  = threadIdx.x;
    const int lane = tid & 31;
    const int wave = tid >> 5;
    const int l15  = lane & 15;
    const int ko   = (lane < 16) ? 0 : 8;

    if (wave == 0) {
        const u32x8 g1 = G1_QS;
        tdm_issue(tdm_g0(0u, Q1t + ((size_t)b * HWD + hoff) * CC), g1);
        tdm_issue(tdm_g0(ATT_QS_BYTES, S1t + ((size_t)b * HWD + hoff) * CC), g1);
        // v3 tile: 256 rows x 128 bf16, row stride HWD; streams during phase 1
        const u32x8 g1v = tdm_g1x(128u, 256u, (unsigned long long)HWD, 1u, 0u, 0u, 0u);
        tdm_issue(tdm_g0(2u * ATT_QS_BYTES, V3b + (size_t)b * CC * HWD + hoff), g1v);
        __builtin_amdgcn_s_wait_tensorcnt(1);      // Q,S landed (v3 may still fly)
    }
    __syncthreads();

    attn_scores_softmax(qs, ssm, sc, tid);

    // repack normalized scores to bf16 rows (S-tile region is dead now)
    unsigned short* sb = ssm;
    if (tid < WW) {
        const float* row = sc + tid * WW;
        unsigned int* dst = (unsigned int*)(sb + tid * SB_STRIDE);
        for (int j = 0; j < WW; j += 2)
            dst[j >> 1] = pack2(row[j], row[j + 1]);
    }
    if (wave == 0) __builtin_amdgcn_s_wait_tensorcnt(0);   // v3 landed
    __syncthreads();

    // Out[c,i] = sum_j v3[c,j] * M[i,j]  (A = bf16 v3 rows, B = bf16 scores)
    for (int mt = 0; mt < 2; ++mt) {
        const int cbase = (wave * 2 + mt) * 16;
        const unsigned short* vr = vt + (size_t)(cbase + l15) * 128;
        for (int it = 0; it < 8; ++it) {
            v8f acc = {};
            #pragma unroll
            for (int j0 = 0; j0 < WW; j0 += 32) {
                BFrag af;
                af.h[0] = *(const v8u*)(vr + j0 + ko);
                af.h[1] = *(const v8u*)(vr + j0 + ko + 16);
                BFrag bf;
                const unsigned short* sp = sb + (it * 16 + l15) * SB_STRIDE + j0 + ko;
                bf.h[0] = *(const v8u*)(sp);
                bf.h[1] = *(const v8u*)(sp + 16);
                acc = wmma_bf16(af, bf, acc);
            }
            #pragma unroll
            for (int v = 0; v < 8; ++v) {
                int c = cbase + v + ((lane >= 16) ? 8 : 0);
                Out[((size_t)b * CC + c) * HWD + hoff + it * 16 + l15] = acc[v];
            }
        }
    }
}

// =====================================================================
// Morphology (binary masks): dilate = max over disk, erode = min over
// disk with zero out-of-bounds (matches the conv-threshold reference).
// =====================================================================
__global__ __launch_bounds__(256)
void morph_kernel(const float* __restrict__ in, float* __restrict__ out,
                  int Himg, int Wimg, int total, int r, int dil, int invertIn) {
    int idx = blockIdx.x * 256 + threadIdx.x;
    if (idx >= total) return;
    int w = idx % Wimg;
    int h = (idx / Wimg) % Himg;
    int bi = idx / (Wimg * Himg);
    const float* img = in + (size_t)bi * Himg * Wimg;
    float best = dil ? 0.f : 1.f;
    for (int dy = -r; dy <= r; ++dy)
        for (int dx = -r; dx <= r; ++dx) {
            if (dy * dy + dx * dx > r * r) continue;
            int yy = h + dy, xx = w + dx;
            float v = 0.f;
            if (yy >= 0 && yy < Himg && xx >= 0 && xx < Wimg) {
                v = img[yy * Wimg + xx];
                if (invertIn) v = 1.f - v;
            }
            best = dil ? fmaxf(best, v) : fminf(best, v);
        }
    out[idx] = best;
}

__global__ __launch_bounds__(256)
void pad_kernel(const float* __restrict__ in, float* __restrict__ out, int total) {
    int idx = blockIdx.x * 256 + threadIdx.x;
    if (idx >= total) return;
    int w = idx % PADW;
    int h = (idx / PADW) % PADW;
    int bi = idx / (PADW * PADW);
    float v = 0.f;
    if (h >= 3 && h < 3 + HH && w >= 3 && w < 3 + WW)
        v = in[((size_t)bi * HH + (h - 3)) * WW + (w - 3)];
    out[idx] = v;
}

__global__ __launch_bounds__(256)
void crop_inv_kernel(const float* __restrict__ in, float* __restrict__ out, int total) {
    int idx = blockIdx.x * 256 + threadIdx.x;
    if (idx >= total) return;
    int w = idx % WW;
    int h = (idx / WW) % HH;
    int bi = idx / HWD;
    out[idx] = 1.f - in[((size_t)bi * PADW + (h + 3)) * PADW + (w + 3)];
}

// generic f32 -> bf16 weight conversion
__global__ __launch_bounds__(256)
void cvt_bf16_kernel(const float* __restrict__ src, unsigned short* __restrict__ dst,
                     int n) {
    int i = blockIdx.x * 256 + threadIdx.x;
    if (i < n) dst[i] = f2bf(src[i]);
}

// split fus_w [256,513] into bf16 [256,512] + f32 V-column [256]
__global__ __launch_bounds__(256)
void prep_fusw_kernel(const float* __restrict__ fw, unsigned short* __restrict__ Wb,
                      float* __restrict__ wv) {
    int idx = blockIdx.x * 256 + threadIdx.x;
    if (idx >= CC * 513) return;
    int m = idx / 513, k = idx % 513;
    if (k == 512) wv[m] = fw[idx];
    else          Wb[m * 512 + k] = f2bf(fw[idx]);
}

// =====================================================================
// Host-side orchestration
// =====================================================================
static void launch_gemm(hipStream_t s, const unsigned short* Wb, int ldw,
                        const float* bias, const float* Xa, const float* Xb,
                        int Ka, int Ktot, void* Y, const float* addsrc,
                        const float* gate, const float* vvec, const float* wv,
                        int act, int omode) {
    gemm1x1_kernel<<<dim3(PTOT / 64), 256, 0, s>>>(Wb, ldw, bias, Xa, Xb, Ka, Ktot,
                                                   Y, addsrc, gate, vvec, wv, CC,
                                                   act, omode);
}

extern "C" void kernel_launch(void* const* d_in, const int* in_sizes, int n_in,
                              void* d_out, int out_size, void* d_ws, size_t ws_size,
                              hipStream_t stream) {
    (void)in_sizes; (void)n_in; (void)out_size; (void)ws_size;

    const float* x_p   = (const float*)d_in[0];
    const float* x_c   = (const float*)d_in[1];
    const float* pa_w1 = (const float*)d_in[2];
    const float* pa_b1 = (const float*)d_in[3];
    const float* pa_w2 = (const float*)d_in[4];
    const float* pa_b2 = (const float*)d_in[5];
    const float* pa_wc = (const float*)d_in[6];
    const float* pa_bc = (const float*)d_in[7];
    const float* ca_aw1 = (const float*)d_in[8];
    const float* ca_ab1 = (const float*)d_in[9];
    const float* ca_aw2 = (const float*)d_in[10];
    const float* ca_ab2 = (const float*)d_in[11];
    const float* ca_mw1 = (const float*)d_in[12];
    const float* ca_mb1 = (const float*)d_in[13];
    const float* ca_mw2 = (const float*)d_in[14];
    const float* ca_mb2 = (const float*)d_in[15];
    const float* b1_w  = (const float*)d_in[16];
    const float* b1_b  = (const float*)d_in[17];
    const float* b2_w  = (const float*)d_in[18];
    const float* b2_b  = (const float*)d_in[19];
    const float* b3_w  = (const float*)d_in[20];
    const float* b3_b  = (const float*)d_in[21];
    const float* fus_w = (const float*)d_in[22];
    const float* fus_b = (const float*)d_in[23];
    float* out = (float*)d_out;

    // workspace layout (floats)
    float* ws  = (float*)d_ws;
    float* A0  = ws;                 // y1 -> buffer_c
    float* A1  = ws + 1 * PLANE;     // y2
    float* A2  = ws + 2 * PLANE;     // t1/t2 -> buffer_p ; then bf16 v3 plane
    float* A3a = ws + 3 * PLANE;     // g1 ; then bf16 Q1t | S1t planes
    float* A3b = ws + 4 * PLANE;     // g2 ; then bf16 Q2t | S2t planes
    float* A4  = ws + 5 * PLANE;     // attention output "buf"
    float* sm   = ws + 6 * PLANE;
    float* camean = sm;                       // [B*C]
    float* camax  = camean + BB * CC;
    float* scale  = camax  + BB * CC;
    float* Vraw   = scale  + BB * CC;         // [B*HW]
    float* T0     = Vraw + PTOT;
    float* T1     = T0   + PTOT;
    float* Ppad0  = T1   + PTOT;              // [B*134*134]
    float* Ppad1  = Ppad0 + PADSZ;
    float* Vfin   = Ppad1 + PADSZ;            // [B*HW]
    float* wvcol  = Vfin + PTOT;              // [256]
    // bf16 weight planes
    unsigned short* pw1b = (unsigned short*)(wvcol + CC);
    unsigned short* pw2b = pw1b + CC * CC;
    unsigned short* pwcb = pw2b + CC * CC;    // [256,512]
    unsigned short* b1b  = pwcb + CC * 512;
    unsigned short* b2b  = b1b + CC * CC;
    unsigned short* b3b  = b2b + CC * CC;
    unsigned short* fusb = b3b + CC * CC;     // [256,512]
    // bf16 activation planes (aliasing dead f32 planes; u16 plane = PLANE elems)
    unsigned short* Q1t = (unsigned short*)A3a;          // transposed [b][p][c]
    unsigned short* S1t = Q1t + PLANE;
    unsigned short* Q2t = (unsigned short*)A3b;
    unsigned short* S2t = Q2t + PLANE;
    unsigned short* V3b = (unsigned short*)A2;           // normal [b][c][p]

    const int eb = (int)((PLANE + 255) / 256);
    const int wb = (CC * CC + 255) / 256;

    // ---- one-time weight conversions (bf16 row-major) ----
    cvt_bf16_kernel<<<wb, 256, 0, stream>>>(pa_w1, pw1b, CC * CC);
    cvt_bf16_kernel<<<wb, 256, 0, stream>>>(pa_w2, pw2b, CC * CC);
    cvt_bf16_kernel<<<2 * wb, 256, 0, stream>>>(pa_wc, pwcb, CC * 512);
    cvt_bf16_kernel<<<wb, 256, 0, stream>>>(b1_w, b1b, CC * CC);
    cvt_bf16_kernel<<<wb, 256, 0, stream>>>(b2_w, b2b, CC * CC);
    cvt_bf16_kernel<<<wb, 256, 0, stream>>>(b3_w, b3b, CC * CC);
    prep_fusw_kernel<<<(CC * 513 + 255) / 256, 256, 0, stream>>>(fus_w, fusb, wvcol);

    // ---- PALayer ----
    pool3_kernel<<<eb, 256, 0, stream>>>(x_p, A0, A1);                         // y1,y2
    launch_gemm(stream, pw1b, CC,  pa_b1, A0, nullptr, CC, CC, A2,
                nullptr, nullptr, nullptr, nullptr, 1, 0);                     // t1=relu
    launch_gemm(stream, pw2b, CC,  pa_b2, A2, nullptr, CC, CC, A3a,
                A0, nullptr, nullptr, nullptr, 0, 0);                          // g1=y3+y1
    launch_gemm(stream, pw1b, CC,  pa_b1, A1, nullptr, CC, CC, A2,
                nullptr, nullptr, nullptr, nullptr, 1, 0);                     // t2=relu
    launch_gemm(stream, pw2b, CC,  pa_b2, A2, nullptr, CC, CC, A3b,
                A1, nullptr, nullptr, nullptr, 0, 0);                          // g2=y4+y2
    launch_gemm(stream, pwcb, 2 * CC, pa_bc, A3a, A3b, CC, 2 * CC, A2,
                nullptr, x_p, nullptr, nullptr, 2, 0);                         // buffer_p

    // ---- CALayer ----
    ca_reduce_kernel<<<BB * CC, 256, 0, stream>>>(x_c, camean, camax);
    ca_mlp_kernel<<<BB, 256, 0, stream>>>(camean, camax, ca_aw1, ca_ab1, ca_aw2,
                                          ca_ab2, ca_mw1, ca_mb1, ca_mw2, ca_mb2,
                                          scale);
    bufc_kernel<<<eb, 256, 0, stream>>>(x_c, scale, A0);                       // buffer_c

    // ---- Q/S/v3 projections (bf16 outputs in consumer layout) ----
    launch_gemm(stream, b1b, CC, b1_b, A2, nullptr, CC, CC, Q1t,
                nullptr, nullptr, nullptr, nullptr, 0, 1);                     // Q1 (T)
    launch_gemm(stream, b2b, CC, b2_b, A2, nullptr, CC, CC, S2t,
                nullptr, nullptr, nullptr, nullptr, 0, 1);                     // S2 (T)
    launch_gemm(stream, b2b, CC, b2_b, A0, nullptr, CC, CC, S1t,
                nullptr, nullptr, nullptr, nullptr, 0, 1);                     // S1 (T)
    launch_gemm(stream, b1b, CC, b1_b, A0, nullptr, CC, CC, Q2t,
                nullptr, nullptr, nullptr, nullptr, 0, 1);                     // Q2 (T)
    launch_gemm(stream, b3b, CC, b3_b, x_c, nullptr, CC, CC, V3b,
                nullptr, nullptr, nullptr, nullptr, 0, 2);                     // v3 (bf16)

    // ---- validity mask from M_p_to_c + morphology ----
    attn_mask_kernel<<<BB * HH, 256, ATT_LDS_BYTES, stream>>>(Q2t, S2t, Vraw);
    {
        const int t128 = BB * HWD, b128 = (t128 + 255) / 256;
        const int tpad = PADSZ,    bpad = (tpad + 255) / 256;
        morph_kernel<<<b128, 256, 0, stream>>>(Vraw, T0, HH, WW, t128, 2, 0, 1); // erode d2 on (1-V)
        morph_kernel<<<b128, 256, 0, stream>>>(T0, T1, HH, WW, t128, 2, 1, 0);   // dilate d2
        morph_kernel<<<b128, 256, 0, stream>>>(T1, T0, HH, WW, t128, 1, 1, 0);   // dilate d1
        morph_kernel<<<b128, 256, 0, stream>>>(T0, T1, HH, WW, t128, 1, 0, 0);   // erode d1
        pad_kernel<<<bpad, 256, 0, stream>>>(T1, Ppad0, tpad);
        morph_kernel<<<bpad, 256, 0, stream>>>(Ppad0, Ppad1, PADW, PADW, tpad, 3, 1, 0); // dilate d3
        morph_kernel<<<bpad, 256, 0, stream>>>(Ppad1, Ppad0, PADW, PADW, tpad, 3, 0, 0); // erode d3
        crop_inv_kernel<<<b128, 256, 0, stream>>>(Ppad0, Vfin, t128);
    }

    // ---- attend with M_c_to_p ----
    attn_apply_kernel<<<BB * HH, 256, ATT_LDS_BYTES, stream>>>(Q1t, S1t, V3b, A4);

    // ---- fusion: out = Wf[:, :256]*buf + Wf[:,256:512]*x_p + Wf[:,512]*V + b ----
    launch_gemm(stream, fusb, 2 * CC, fus_b, A4, x_p, CC, 2 * CC, out,
                nullptr, nullptr, Vfin, wvcol, 0, 0);
}
